// ParallelExperts_67860483277374
// MI455X (gfx1250) — matrix operand order repaired and checked by reading, not compiled, not run on hardware
//
#include <hip/hip_runtime.h>
#include <hip/hip_bf16.h>
#include <math.h>
#include <stdint.h>

// ---------------- problem constants (from reference) ----------------
#define E      8
#define D_IN   1024
#define D_HID  4096
#define D_OUT  1024
#define N_TOK  32768
#define CAP    (N_TOK / E)     // 4096 tokens per expert

// ---------------- tiling ----------------
#define BM 128
#define BN 128
#define BK 64
#define LDS_STRIDE 72          // ushorts per LDS row (144B: 16B aligned, de-conflicted)

typedef __attribute__((ext_vector_type(16))) __bf16 v16bf;
typedef __attribute__((ext_vector_type(8)))  float  v8f;
typedef __attribute__((ext_vector_type(4)))  int    v4i;

#define AS_GLOBAL __attribute__((address_space(1)))
#define AS_LOCAL  __attribute__((address_space(3)))

// fp32 -> bf16 bits via the hardware convert (RNE); clang lowers the scalar
// cast to v_cvt_(pk_)bf16_f32 on gfx1250.
__device__ __forceinline__ unsigned short f2bf(float f) {
    union { __bf16 h; unsigned short u; } c;
    c.h = (__bf16)f;
    return c.u;
}

// Branch-free exact-erf GELU: gelu(x) = 0.5 x (1 + erf(x/sqrt(2))).
// erf via Abramowitz–Stegun 7.1.26 rational approx, |err| <= 1.5e-7
// (negligible vs the bf16 quantization of the operands). No divergence.
__device__ __forceinline__ float gelu_erf(float x) {
    const float ax = fabsf(x) * 0.70710678118654752440f;
    const float t  = __builtin_amdgcn_rcpf(1.0f + 0.3275911f * ax);
    float p = 1.061405429f;
    p = p * t - 1.453152027f;
    p = p * t + 1.421413741f;
    p = p * t - 0.284496736f;
    p = p * t + 0.254829592f;
    const float er = 1.0f - (p * t) * __expf(-ax * ax);   // erf(|x|/sqrt2)
    const float erf_signed = copysignf(er, x);
    return 0.5f * x * (1.0f + erf_signed);
}

// A-matrix 16x32 bf16 fragment (ISA 7.12.2): lanes 0-15 hold K=0..7 & 16..23,
// lanes 16-31 hold K=8..15 & 24..31. Two 16B chunks, 32B apart.
__device__ __forceinline__ v16bf load_fragA(const unsigned short* row, int kk, int lane) {
    union { v16bf v; uint4 q[2]; } u;
    const int klo = kk + ((lane >> 4) << 3);           // +8 for upper half-wave
    u.q[0] = *(const uint4*)(row + klo);
    u.q[1] = *(const uint4*)(row + klo + 16);
    return u.v;
}

// B-matrix 32x16 bf16 fragment: lane = column N; lanes 0-15 hold K=0..7 & 16..23,
// lanes 16-31 hold K=8..15 & 24..31. Two 16B chunks, 32B apart.
__device__ __forceinline__ v16bf load_fragB(const unsigned short* row, int kk, int lane) {
    union { v16bf v; uint4 q[2]; } u;
    const int klo = kk + ((lane >> 4) << 3);           // +8 for upper half-wave
    u.q[0] = *(const uint4*)(row + klo);
    u.q[1] = *(const uint4*)(row + klo + 16);
    return u.v;
}

// ================= GEMM1: H = gelu_erf(X @ W1 + b1), H stored as bf16 =================
__global__ __launch_bounds__(256) void moe_gemm1_gelu(
    const float* __restrict__ X,    // [E*CAP, D_IN] fp32
    const float* __restrict__ W1,   // [E, D_IN, D_HID] fp32
    const float* __restrict__ B1,   // [E, D_HID] fp32
    unsigned short* __restrict__ H) // [E*CAP, D_HID] bf16 (workspace)
{
    __shared__ unsigned short lA[BM * LDS_STRIDE];
    __shared__ unsigned short lB[BN * LDS_STRIDE];

    const int e   = blockIdx.z;
    const int m0  = blockIdx.y * BM;
    const int n0  = blockIdx.x * BN;
    const int tid = threadIdx.x;
    const int lane = tid & 31;
    const int wid  = tid >> 5;
    const int wm   = (wid >> 2) * 64;   // wave row offset: 0 / 64
    const int wn   = (wid & 3) * 32;    // wave col offset: 0..96

    const float* Xe = X  + (size_t)e * CAP  * D_IN;
    const float* We = W1 + (size_t)e * D_IN * D_HID;

    v8f acc[4][2];
#pragma unroll
    for (int mt = 0; mt < 4; ++mt)
#pragma unroll
        for (int nt = 0; nt < 2; ++nt)
            acc[mt][nt] = (v8f){0.f,0.f,0.f,0.f,0.f,0.f,0.f,0.f};

    // staging index maps
    const int ar = tid >> 1;            // A: row 0..127
    const int ac = (tid & 1) * 32;      // A: k half 0/32
    const int bk = tid & 63;            // B: k row 0..63
    const int bn = (tid >> 6) * 32;     // B: n group base (0/32/64/96)

    for (int k0 = 0; k0 < D_IN; k0 += BK) {
        __syncthreads();
        // ---- stage A tile: X[m0..+128, k0..+64] -> bf16 LDS [m][k]
        {
            const float* xrow = Xe + (size_t)(m0 + ar) * D_IN + k0 + ac;
#pragma unroll
            for (int j = 0; j < 8; ++j) {
                float4 f = *(const float4*)(xrow + j * 4);
                uint2 p;
                p.x = (unsigned)f2bf(f.x) | ((unsigned)f2bf(f.y) << 16);
                p.y = (unsigned)f2bf(f.z) | ((unsigned)f2bf(f.w) << 16);
                *(uint2*)&lA[ar * LDS_STRIDE + ac + j * 4] = p;
            }
        }
        // ---- stage B tile: W1[k0..+64, n0..+128] -> bf16 LDS transposed [n][k]
        {
            const float* wrow = We + (size_t)(k0 + bk) * D_HID + n0 + bn;
#pragma unroll
            for (int j = 0; j < 8; ++j) {
                float4 f = *(const float4*)(wrow + j * 4);
                const int nb = bn + j * 4;
                lB[(nb + 0) * LDS_STRIDE + bk] = f2bf(f.x);
                lB[(nb + 1) * LDS_STRIDE + bk] = f2bf(f.y);
                lB[(nb + 2) * LDS_STRIDE + bk] = f2bf(f.z);
                lB[(nb + 3) * LDS_STRIDE + bk] = f2bf(f.w);
            }
        }
        __syncthreads();

#pragma unroll
        for (int kk = 0; kk < BK; kk += 32) {
            // Load ALL fragments first, then issue the 8 WMMAs back-to-back.
            v16bf afr[4], bfr[2];
#pragma unroll
            for (int nt = 0; nt < 2; ++nt)
                bfr[nt] = load_fragB(&lB[(wn + nt * 16 + (lane & 15)) * LDS_STRIDE], kk, lane);
#pragma unroll
            for (int mt = 0; mt < 4; ++mt)
                afr[mt] = load_fragA(&lA[(wm + mt * 16 + (lane & 15)) * LDS_STRIDE], kk, lane);
#pragma unroll
            for (int mt = 0; mt < 4; ++mt)
#pragma unroll
                for (int nt = 0; nt < 2; ++nt)
                    acc[mt][nt] = __builtin_amdgcn_wmma_f32_16x16x32_bf16(
                        false, afr[mt], false, bfr[nt], (short)0, acc[mt][nt], false, false);
        }
    }

    // ---- epilogue: bias + branch-free erf GELU, write bf16 H
    const int lm = (lane >> 4) << 3;    // C/D: +8 rows for upper half-wave
    const int ln = lane & 15;           // C/D: N = lane & 15
#pragma unroll
    for (int nt = 0; nt < 2; ++nt) {
        const int ng = n0 + wn + nt * 16 + ln;
        const float bias = B1[(size_t)e * D_HID + ng];
#pragma unroll
        for (int mt = 0; mt < 4; ++mt) {
            const int mbase = m0 + wm + mt * 16 + lm;
#pragma unroll
            for (int r = 0; r < 8; ++r) {
                float g = gelu_erf(acc[mt][nt][r] + bias);
                H[((size_t)e * CAP + (mbase + r)) * D_HID + ng] = f2bf(g);
            }
        }
    }
}

// ================= GEMM2: Y = H(bf16) @ W2 + b2, fp32 out =================
__global__ __launch_bounds__(256) void moe_gemm2(
    const unsigned short* __restrict__ H,  // [E*CAP, D_HID] bf16
    const float* __restrict__ W2,          // [E, D_HID, D_OUT] fp32
    const float* __restrict__ B2,          // [E, D_OUT] fp32
    float* __restrict__ Y)                 // [E*CAP, D_OUT] fp32
{
    __shared__ unsigned short lA[BM * LDS_STRIDE];
    __shared__ unsigned short lB[BN * LDS_STRIDE];

    const int e   = blockIdx.z;
    const int m0  = blockIdx.y * BM;
    const int n0  = blockIdx.x * BN;
    const int tid = threadIdx.x;
    const int lane = tid & 31;
    const int wid  = tid >> 5;
    const int wm   = (wid >> 2) * 64;
    const int wn   = (wid & 3) * 32;

    const unsigned short* He = H  + (size_t)e * CAP   * D_HID;
    const float*          We = W2 + (size_t)e * D_HID * D_OUT;

    v8f acc[4][2];
#pragma unroll
    for (int mt = 0; mt < 4; ++mt)
#pragma unroll
        for (int nt = 0; nt < 2; ++nt)
            acc[mt][nt] = (v8f){0.f,0.f,0.f,0.f,0.f,0.f,0.f,0.f};

    const int ar = tid >> 1;
    const int ac = (tid & 1) * 32;
    const int bk = tid & 63;
    const int bn = (tid >> 6) * 32;

    for (int k0 = 0; k0 < D_HID; k0 += BK) {
        __syncthreads();
        // ---- stage A tile: H already bf16 -> pure global->LDS copy.
        // Prefer the CDNA5 async path (no VGPR round trip, ASYNCcnt tracked).
        {
            const unsigned short* hrow = He + (size_t)(m0 + ar) * D_HID + k0 + ac;
            unsigned short* lrow = &lA[ar * LDS_STRIDE + ac];
#if __has_builtin(__builtin_amdgcn_global_load_async_to_lds_b128)
#pragma unroll
            for (int j = 0; j < 4; ++j) {
                AS_GLOBAL v4i* g = (AS_GLOBAL v4i*)(uintptr_t)(const void*)(hrow + j * 8);
                AS_LOCAL  v4i* l = (AS_LOCAL  v4i*)(uintptr_t)(void*)(lrow + j * 8);
                __builtin_amdgcn_global_load_async_to_lds_b128(g, l, 0, 0);
            }
#if __has_builtin(__builtin_amdgcn_s_wait_asynccnt)
            __builtin_amdgcn_s_wait_asynccnt(0);
#else
            asm volatile("s_wait_asynccnt 0" ::: "memory");
#endif
#else
#pragma unroll
            for (int j = 0; j < 4; ++j)
                *(uint4*)(lrow + j * 8) = *(const uint4*)(hrow + j * 8);
#endif
        }
        // ---- stage B tile: W2[k0..+64, n0..+128] -> bf16 LDS transposed [n][k]
        {
            const float* wrow = We + (size_t)(k0 + bk) * D_OUT + n0 + bn;
#pragma unroll
            for (int j = 0; j < 8; ++j) {
                float4 f = *(const float4*)(wrow + j * 4);
                const int nb = bn + j * 4;
                lB[(nb + 0) * LDS_STRIDE + bk] = f2bf(f.x);
                lB[(nb + 1) * LDS_STRIDE + bk] = f2bf(f.y);
                lB[(nb + 2) * LDS_STRIDE + bk] = f2bf(f.z);
                lB[(nb + 3) * LDS_STRIDE + bk] = f2bf(f.w);
            }
        }
        __syncthreads();

#pragma unroll
        for (int kk = 0; kk < BK; kk += 32) {
            v16bf afr[4], bfr[2];
#pragma unroll
            for (int nt = 0; nt < 2; ++nt)
                bfr[nt] = load_fragB(&lB[(wn + nt * 16 + (lane & 15)) * LDS_STRIDE], kk, lane);
#pragma unroll
            for (int mt = 0; mt < 4; ++mt)
                afr[mt] = load_fragA(&lA[(wm + mt * 16 + (lane & 15)) * LDS_STRIDE], kk, lane);
#pragma unroll
            for (int mt = 0; mt < 4; ++mt)
#pragma unroll
                for (int nt = 0; nt < 2; ++nt)
                    acc[mt][nt] = __builtin_amdgcn_wmma_f32_16x16x32_bf16(
                        false, afr[mt], false, bfr[nt], (short)0, acc[mt][nt], false, false);
        }
    }

    const int lm = (lane >> 4) << 3;
    const int ln = lane & 15;
#pragma unroll
    for (int nt = 0; nt < 2; ++nt) {
        const int ng = n0 + wn + nt * 16 + ln;
        const float bias = B2[(size_t)e * D_OUT + ng];
#pragma unroll
        for (int mt = 0; mt < 4; ++mt) {
            const int mbase = m0 + wm + mt * 16 + lm;
#pragma unroll
            for (int r = 0; r < 8; ++r)
                Y[((size_t)e * CAP + (mbase + r)) * D_OUT + ng] = acc[mt][nt][r] + bias;
        }
    }
}

extern "C" void kernel_launch(void* const* d_in, const int* in_sizes, int n_in,
                              void* d_out, int out_size, void* d_ws, size_t ws_size,
                              hipStream_t stream) {
    (void)in_sizes; (void)n_in; (void)out_size; (void)ws_size;
    const float* X  = (const float*)d_in[0];
    // d_in[1] = expert_size (int64): equal splits by construction, unused
    const float* W1 = (const float*)d_in[2];
    const float* B1 = (const float*)d_in[3];
    const float* W2 = (const float*)d_in[4];
    const float* B2 = (const float*)d_in[5];
    float* Y = (float*)d_out;
    unsigned short* H = (unsigned short*)d_ws;   // bf16 intermediate, 32768*4096*2 = 256 MB

    dim3 blk(256);
    moe_gemm1_gelu<<<dim3(D_HID / BN, CAP / BM, E), blk, 0, stream>>>(X, W1, B1, H);
    moe_gemm2    <<<dim3(D_OUT / BN, CAP / BM, E), blk, 0, stream>>>(H, W2, B2, Y);
}
